// HGSL_4870492913804
// MI455X (gfx1250) — matrix operand order repaired and checked by reading, not compiled, not run on hardware
//
#include <hip/hip_runtime.h>

typedef __attribute__((ext_vector_type(16))) _Float16 v16h;
typedef __attribute__((ext_vector_type(8)))  float    v8f;

#define DEV __device__ __forceinline__

// ---------------------------------------------------------------------------
// WMMA helpers (gfx1250, wave32). Layouts per CDNA5 ISA 7.12.2:
//  A (16x32 f16): lane = {m=lane&15, kgrp=lane>>4}; half h -> k = kgrp*8 + (h<8 ? h : h+8)
//  B (32x16 f16): lane = {n=lane&15, kgrp=lane>>4}; half h -> k = kgrp*8 + (h<8 ? h : h+8)
//  C/D (16x16 f32): lane = {n=lane&15}; vgpr r -> m = (lane>>4)*8 + r
// B operands are pre-swizzled to f16 so each tile is one contiguous 32B
// per-lane load (2x global_load_b128) instead of 16 scalar loads.
// ---------------------------------------------------------------------------
DEV v8f vzero8() {
  v8f z;
#pragma unroll
  for (int i = 0; i < 8; ++i) z[i] = 0.f;
  return z;
}

DEV v8f wmma16(v16h a, v16h b, v8f c) {
  return __builtin_amdgcn_wmma_f32_16x16x32_f16(false, a, false, b, (short)0, c,
                                                false, false);
}

DEV v16h a_tile_f32(const float* src, int lda, int row0, int k0, int lane) {
  int m  = row0 + (lane & 15);
  int kg = (lane >> 4) * 8;
  v16h a;
#pragma unroll
  for (int h = 0; h < 16; ++h) {
    int k = k0 + kg + ((h < 8) ? h : (h + 8));
    a[h] = (_Float16)src[m * lda + k];
  }
  return a;
}

DEV v16h a_tile_f32g(const float* src, int lda, int row0, int k0, int lane, int nrow) {
  int m  = row0 + (lane & 15);
  int kg = (lane >> 4) * 8;
  bool ok = (m < nrow);
  v16h a;
#pragma unroll
  for (int h = 0; h < 16; ++h) {
    int k = k0 + kg + ((h < 8) ? h : (h + 8));
    a[h] = ok ? (_Float16)src[m * lda + k] : (_Float16)0.f;
  }
  return a;
}

DEV v16h a_tile_f16(const _Float16* src, int lda, int row0, int k0, int lane) {
  int m  = row0 + (lane & 15);
  int kg = (lane >> 4) * 8;
  v16h a;
#pragma unroll
  for (int h = 0; h < 16; ++h) {
    int k = k0 + kg + ((h < 8) ? h : (h + 8));
    a[h] = src[m * lda + k];
  }
  return a;
}

// Pre-swizzled B tile: slot layout [tile(8)][lane(32)][h(16)] halves.
// tile = kstep*4 + nt  (kstep: 0 -> k0=0, 1 -> k0=32).
DEV v16h b_tile_sw(const _Float16* Wsw, int tile, int lane) {
  return *(const v16h*)(Wsw + (((tile << 5) + lane) << 4));
}

DEV void store_c_f16(_Float16* dst, int ldc, int row0, int n0, int lane, v8f c) {
  int n  = n0 + (lane & 15);
  int m0 = row0 + ((lane >> 4) << 3);
#pragma unroll
  for (int r = 0; r < 8; ++r) dst[(m0 + r) * ldc + n] = (_Float16)c[r];
}

// ---------------------------------------------------------------------------
// K0: swizzle all 14 64x64 weight matrices into B-tile-ready f16 layout.
// slots: 0=w1, 1=w2, 2..7=qkv[br*3+proj], 8..9=Wo[br], 10..13=ffW[br*2+layer]
// ---------------------------------------------------------------------------
__global__ void k_prep(const float* __restrict__ w1, const float* __restrict__ w2,
                       const float* __restrict__ qkv, const float* __restrict__ Wo,
                       const float* __restrict__ ffW, _Float16* __restrict__ Wsw) {
  int idx = blockIdx.x * 256 + threadIdx.x;
  if (idx >= 14 * 4096) return;
  int h = idx & 15, lane = (idx >> 4) & 31, tile = (idx >> 9) & 7, mat = idx >> 12;
  int k = ((tile >> 2) << 5) + ((lane >> 4) << 3) + ((h < 8) ? h : (h + 8));
  int n = ((tile & 3) << 4) + (lane & 15);
  const float* src;
  if (mat == 0) src = w1;
  else if (mat == 1) src = w2;
  else if (mat < 8) src = qkv + (mat - 2) * 4096;
  else if (mat < 10) src = Wo + (mat - 8) * 4096;
  else src = ffW + (mat - 10) * 4096;
  Wsw[idx] = (_Float16)src[k * 64 + n];
}

// ---------------------------------------------------------------------------
// K1: xin[n,:] = user_emb[n,:] + global_cen_emb[user_cen[n],:]   (float4)
// ---------------------------------------------------------------------------
__global__ void k_xin(const float4* __restrict__ ue4, const float4* __restrict__ gce4,
                      const int* __restrict__ cen, float4* __restrict__ xin4, int Nn) {
  int i = blockIdx.x * 256 + threadIdx.x;
  if (i >= Nn * 16) return;
  int n = i >> 4, c4 = i & 15;
  float4 a = ue4[i];
  float4 g = gce4[cen[n] * 16 + c4];
  xin4[i] = make_float4(a.x + g.x, a.y + g.y, a.z + g.z, a.w + g.w);
}

// ---------------------------------------------------------------------------
// K2: generic C[M,64] = A[M,64] @ W[64,64] via WMMA f16->f32.
// 128 threads = 4 waves, each wave one 16-row tile. Wave-uniform fast path
// for full tiles (no guards); guarded path only for the single tail block.
// ---------------------------------------------------------------------------
__global__ __launch_bounds__(128)
void k_gemm64(const float* __restrict__ A, const _Float16* __restrict__ Wsw,
              float* __restrict__ C, int M) {
  int lane = threadIdx.x & 31;
  int wv   = threadIdx.x >> 5;
  int row0 = blockIdx.x * 64 + wv * 16;
  if (row0 >= M) return;           // wave-uniform; EXEC all-ones for WMMA
  bool full = (row0 + 16 <= M);    // wave-uniform
  if (full) {
    v16h a0 = a_tile_f32(A, 64, row0, 0, lane);
    v16h a1 = a_tile_f32(A, 64, row0, 32, lane);
#pragma unroll
    for (int nt = 0; nt < 4; ++nt) {
      v8f acc = vzero8();
      acc = wmma16(a0, b_tile_sw(Wsw, nt, lane), acc);
      acc = wmma16(a1, b_tile_sw(Wsw, 4 + nt, lane), acc);
      int n  = nt * 16 + (lane & 15);
      int m0 = row0 + ((lane >> 4) << 3);
#pragma unroll
      for (int r = 0; r < 8; ++r) C[(m0 + r) * 64 + n] = acc[r];
    }
  } else {
    v16h a0 = a_tile_f32g(A, 64, row0, 0, lane, M);
    v16h a1 = a_tile_f32g(A, 64, row0, 32, lane, M);
#pragma unroll
    for (int nt = 0; nt < 4; ++nt) {
      v8f acc = vzero8();
      acc = wmma16(a0, b_tile_sw(Wsw, nt, lane), acc);
      acc = wmma16(a1, b_tile_sw(Wsw, 4 + nt, lane), acc);
      int n  = nt * 16 + (lane & 15);
      int m0 = row0 + ((lane >> 4) << 3);
#pragma unroll
      for (int r = 0; r < 8; ++r) {
        int m = m0 + r;
        if (m < M) C[m * 64 + n] = acc[r];
      }
    }
  }
}

// ---------------------------------------------------------------------------
// K3: edge[e,:] = relu(masked_mean_l xw[seq[e,l],:]);  pad == (idx==0).
// float4 per thread; 16 threads per edge, 16 edges per 256-thread block.
// ---------------------------------------------------------------------------
__global__ __launch_bounds__(256)
void k_edge(const float4* __restrict__ xw4, const int* __restrict__ seq,
            float4* __restrict__ edge4, int E) {
  int t = blockIdx.x * 256 + threadIdx.x;
  if (t >= E * 16) return;
  int e = t >> 4, c4 = t & 15;
  const int* row = seq + e * 50;
  float4 acc = make_float4(0.f, 0.f, 0.f, 0.f);
  int cnt = 0;
  for (int l = 0; l < 50; ++l) {
    int u = row[l];
    if (u > 0) {
      float4 v = xw4[u * 16 + c4];
      acc.x += v.x; acc.y += v.y; acc.z += v.z; acc.w += v.w;
      ++cnt;
    }
  }
  float4 res;
  if (cnt > 0) {
    float inv = 1.f / (float)cnt;
    res = make_float4(acc.x * inv, acc.y * inv, acc.z * inv, acc.w * inv);
  } else {
    res = xw4[c4];  // all-pad: uniform mean of 50 copies of row 0
  }
  edge4[t] = make_float4(fmaxf(res.x, 0.f), fmaxf(res.y, 0.f),
                         fmaxf(res.z, 0.f), fmaxf(res.w, 0.f));
}

// ---------------------------------------------------------------------------
// K4: adj[b,l,:] = masked_mean_m ew2[useq[seq[data_idx[b],l], m], :]  (float4)
// ---------------------------------------------------------------------------
__global__ __launch_bounds__(256)
void k_adj(const float4* __restrict__ ew24, const int* __restrict__ seq,
           const int* __restrict__ data_idx, const int* __restrict__ useq,
           float4* __restrict__ adj4, int BL) {
  int t = blockIdx.x * 256 + threadIdx.x;
  if (t >= BL * 16) return;
  int blk = t >> 4, c4 = t & 15;  // blk = b*50 + l
  int b = blk / 50, l = blk % 50;
  int di = data_idx[b];
  int u  = seq[di * 50 + l];
  const int* row = useq + u * 20;
  float4 acc = make_float4(0.f, 0.f, 0.f, 0.f);
  int cnt = 0;
  for (int m = 0; m < 20; ++m) {
    int e = row[m];
    if (e > 0) {
      float4 v = ew24[e * 16 + c4];
      acc.x += v.x; acc.y += v.y; acc.z += v.z; acc.w += v.w;
      ++cnt;
    }
  }
  float4 res;
  if (cnt > 0) {
    float inv = 1.f / (float)cnt;
    res = make_float4(acc.x * inv, acc.y * inv, acc.z * inv, acc.w * inv);
  } else {
    res = ew24[c4];
  }
  adj4[t] = res;
}

// ---------------------------------------------------------------------------
// K5: fused transformer block + masked-mean pooling.
// grid = (B, 2 branches), 256 threads = 8 waves.  L=50 padded to 64 rows.
// LDS: sX 16K (f32 acts) | sQKV 24K (f16) | sS 10K | sC 8K | seq_b. ~59 KB.
// All six DxD GEMMs per block go through v_wmma_f32_16x16x32_f16 with
// pre-swizzled f16 B operands.
// ---------------------------------------------------------------------------
__global__ __launch_bounds__(256)
void k_tblock(const float* __restrict__ adj, const int* __restrict__ seq,
              const int* __restrict__ data_idx, const int* __restrict__ timestamps,
              const int* __restrict__ user_level, const int* __restrict__ user_inf,
              const float* __restrict__ lt_emb, const float* __restrict__ cp_emb,
              const float* __restrict__ li_emb, const _Float16* __restrict__ Wsw,
              const float* __restrict__ boB, const float* __restrict__ lnP,
              const float* __restrict__ ffB, float* __restrict__ pool, int Bn) {
  __shared__ float    sX[64 * 64];
  __shared__ _Float16 sQKV[3 * 64 * 64];
  __shared__ float    sS[50 * 50];
  __shared__ _Float16 sC[64 * 64];
  __shared__ int      sSeqb[64];

  int tid = threadIdx.x, lane = tid & 31, wv = tid >> 5;
  int b = blockIdx.x, br = blockIdx.y;
  int di = data_idx[b];

  if (tid < 64) sSeqb[tid] = (tid < 50) ? seq[di * 50 + tid] : 0;

  // build branch input h, zero-padded to 64 rows
  for (int idx = tid; idx < 4096; idx += 256) {
    int l = idx >> 6, c = idx & 63;
    float v = 0.f;
    if (l < 50) {
      v = adj[(b * 50 + l) * 64 + c];
      if (br == 0) v += lt_emb[timestamps[di * 50 + l] * 64 + c];
      else
        v += li_emb[user_inf[di * 50 + l] * 64 + c] +
             cp_emb[user_level[di * 50 + l] * 64 + c];
    }
    sX[idx] = v;
  }
  for (int idx = tid; idx < 4096; idx += 256) sC[idx] = (_Float16)0.f;
  __syncthreads();

  // QKV projections: 3 GEMMs x 16 tiles = 48 WMMA tile jobs over 8 waves
  const _Float16* qslot = Wsw + (2 + br * 3) * 4096;
  for (int j = wv; j < 48; j += 8) {
    int proj = j >> 4, tile = j & 15, mt = tile >> 2, nt = tile & 3;
    const _Float16* Wp = qslot + proj * 4096;
    v16h a0 = a_tile_f32(sX, 64, mt * 16, 0, lane);
    v16h a1 = a_tile_f32(sX, 64, mt * 16, 32, lane);
    v8f acc = vzero8();
    acc = wmma16(a0, b_tile_sw(Wp, nt, lane), acc);
    acc = wmma16(a1, b_tile_sw(Wp, 4 + nt, lane), acc);
    store_c_f16(sQKV + proj * 4096, 64, mt * 16, nt * 16, lane, acc);
  }
  __syncthreads();

  // attention, per head (d_h = 8 -> VALU; WMMA K-shape does not fit)
  const float scl = 0.35355339059327373f;  // 1/sqrt(8)
  for (int h = 0; h < 8; ++h) {
    for (int idx = tid; idx < 2500; idx += 256) {
      int i = idx / 50, jj = idx % 50;
      const _Float16* qr = sQKV + i * 64 + h * 8;
      const _Float16* kr = sQKV + 4096 + jj * 64 + h * 8;
      float s = 0.f;
#pragma unroll
      for (int d = 0; d < 8; ++d) s += (float)qr[d] * (float)kr[d];
      s *= scl;
      if (sSeqb[jj] == 0) s = -1e9f;
      sS[idx] = s;
    }
    __syncthreads();
    if (tid < 50) {  // row softmax
      float* row = sS + tid * 50;
      float mx = row[0];
      for (int jj = 1; jj < 50; ++jj) mx = fmaxf(mx, row[jj]);
      float sum = 0.f;
      for (int jj = 0; jj < 50; ++jj) { float e = expf(row[jj] - mx); row[jj] = e; sum += e; }
      float inv = 1.f / sum;
      for (int jj = 0; jj < 50; ++jj) row[jj] *= inv;
    }
    __syncthreads();
    for (int idx = tid; idx < 400; idx += 256) {  // ctx_h = P_h @ V_h
      int i = idx >> 3, d = idx & 7;
      float acc = 0.f;
      for (int jj = 0; jj < 50; ++jj)
        acc += sS[i * 50 + jj] * (float)sQKV[2 * 4096 + jj * 64 + h * 8 + d];
      sC[i * 64 + h * 8 + d] = (_Float16)acc;
    }
    __syncthreads();
  }

  // out-proj + bias + residual -> sX
  const _Float16* WoS = Wsw + (8 + br) * 4096;
  const float* bo = boB + br * 64;
  for (int j = wv; j < 16; j += 8) {
    int mt = j >> 2, nt = j & 3;
    v16h a0 = a_tile_f16(sC, 64, mt * 16, 0, lane);
    v16h a1 = a_tile_f16(sC, 64, mt * 16, 32, lane);
    v8f acc = vzero8();
    acc = wmma16(a0, b_tile_sw(WoS, nt, lane), acc);
    acc = wmma16(a1, b_tile_sw(WoS, 4 + nt, lane), acc);
    int n = nt * 16 + (lane & 15), m0 = mt * 16 + ((lane >> 4) << 3);
#pragma unroll
    for (int r = 0; r < 8; ++r) sX[(m0 + r) * 64 + n] += acc[r] + bo[n];
  }
  __syncthreads();

  // LN1 (thread-per-row); stash f16 copy in sQKV region 0
  {
    const float* g  = lnP + ((br * 2 + 0) * 2 + 0) * 64;
    const float* be = lnP + ((br * 2 + 0) * 2 + 1) * 64;
    if (tid < 64) {
      float* row = sX + tid * 64;
      float m = 0.f;
      for (int c = 0; c < 64; ++c) m += row[c];
      m *= (1.f / 64.f);
      float v = 0.f;
      for (int c = 0; c < 64; ++c) { float d = row[c] - m; v += d * d; }
      v *= (1.f / 64.f);
      float inv = rsqrtf(v + 1e-5f);
      for (int c = 0; c < 64; ++c) {
        float y = (row[c] - m) * inv * g[c] + be[c];
        row[c] = y;
        sQKV[tid * 64 + c] = (_Float16)y;
      }
    }
  }
  __syncthreads();

  // FF1: relu(x @ W1 + b1) -> f16 into sQKV region 1
  const _Float16* W1S = Wsw + (10 + br * 2 + 0) * 4096;
  const _Float16* W2S = Wsw + (10 + br * 2 + 1) * 4096;
  const float* b1 = ffB + (br * 2 + 0) * 64;
  const float* b2 = ffB + (br * 2 + 1) * 64;
  for (int j = wv; j < 16; j += 8) {
    int mt = j >> 2, nt = j & 3;
    v16h a0 = a_tile_f16(sQKV, 64, mt * 16, 0, lane);
    v16h a1 = a_tile_f16(sQKV, 64, mt * 16, 32, lane);
    v8f acc = vzero8();
    acc = wmma16(a0, b_tile_sw(W1S, nt, lane), acc);
    acc = wmma16(a1, b_tile_sw(W1S, 4 + nt, lane), acc);
    int n = nt * 16 + (lane & 15), m0 = mt * 16 + ((lane >> 4) << 3);
#pragma unroll
    for (int r = 0; r < 8; ++r)
      sQKV[4096 + (m0 + r) * 64 + n] = (_Float16)fmaxf(acc[r] + b1[n], 0.f);
  }
  __syncthreads();

  // FF2 + bias + residual -> sX
  for (int j = wv; j < 16; j += 8) {
    int mt = j >> 2, nt = j & 3;
    v16h a0 = a_tile_f16(sQKV + 4096, 64, mt * 16, 0, lane);
    v16h a1 = a_tile_f16(sQKV + 4096, 64, mt * 16, 32, lane);
    v8f acc = vzero8();
    acc = wmma16(a0, b_tile_sw(W2S, nt, lane), acc);
    acc = wmma16(a1, b_tile_sw(W2S, 4 + nt, lane), acc);
    int n = nt * 16 + (lane & 15), m0 = mt * 16 + ((lane >> 4) << 3);
#pragma unroll
    for (int r = 0; r < 8; ++r) sX[(m0 + r) * 64 + n] += acc[r] + b2[n];
  }
  __syncthreads();

  // LN2
  {
    const float* g  = lnP + ((br * 2 + 1) * 2 + 0) * 64;
    const float* be = lnP + ((br * 2 + 1) * 2 + 1) * 64;
    if (tid < 64) {
      float* row = sX + tid * 64;
      float m = 0.f;
      for (int c = 0; c < 64; ++c) m += row[c];
      m *= (1.f / 64.f);
      float v = 0.f;
      for (int c = 0; c < 64; ++c) { float d = row[c] - m; v += d * d; }
      v *= (1.f / 64.f);
      float inv = rsqrtf(v + 1e-5f);
      for (int c = 0; c < 64; ++c) row[c] = (row[c] - m) * inv * g[c] + be[c];
    }
  }
  __syncthreads();

  // masked-mean pooling over L
  if (tid < 64) {
    int k = 0;
    for (int l = 0; l < 50; ++l) k += (sSeqb[l] > 0) ? 1 : 0;
    float acc = 0.f;
    if (k > 0) {
      for (int l = 0; l < 50; ++l)
        if (sSeqb[l] > 0) acc += sX[l * 64 + tid];
      acc /= (float)k;
    } else {
      for (int l = 0; l < 50; ++l) acc += sX[l * 64 + tid];
      acc *= (1.f / 50.f);
    }
    pool[(br * Bn + b) * 64 + tid] = acc;
  }
}

// ---------------------------------------------------------------------------
// K6: head — concat extras, @weight/@weight2, gated fusion, log-softmax.
// ---------------------------------------------------------------------------
__global__ __launch_bounds__(128)
void k_head(const float* __restrict__ pool, const int* __restrict__ data_idx,
            const float* __restrict__ spread, const float* __restrict__ Wt,
            const float* __restrict__ Ws, const float* __restrict__ l1W,
            const float* __restrict__ l1b, const float* __restrict__ l2W,
            const float* __restrict__ l2b, const float* __restrict__ outW,
            const float* __restrict__ outb, float* __restrict__ out, int Bn) {
  __shared__ float e0[66], e1[66], t0[66], t1[66], u0[66], u1[66];
  __shared__ float sc[2];
  int tid = threadIdx.x, b = blockIdx.x;
  int di = data_idx[b];
  if (tid < 64) {
    e0[tid] = pool[b * 64 + tid];
    e1[tid] = pool[(Bn + b) * 64 + tid];
  }
  if (tid == 64) { e0[64] = spread[di * 4 + 2] * (1.f / 86400.f); e1[64] = spread[di * 4 + 0]; }
  if (tid == 65) { e0[65] = spread[di * 4 + 3] * (1.f / 86400.f); e1[65] = spread[di * 4 + 1]; }
  __syncthreads();
  if (tid < 66) {  // n_t = e0 @ weight ; n_s = e1 @ weight2
    float a0 = 0.f, a1 = 0.f;
    for (int i = 0; i < 66; ++i) { a0 += e0[i] * Wt[i * 66 + tid]; a1 += e1[i] * Ws[i * 66 + tid]; }
    t0[tid] = a0;
    t1[tid] = a1;
  }
  __syncthreads();
  if (tid < 66) {  // tanh(emb @ l1 + b1)
    float a0 = l1b[tid], a1 = l1b[tid];
    for (int i = 0; i < 66; ++i) { a0 += t0[i] * l1W[i * 66 + tid]; a1 += t1[i] * l1W[i * 66 + tid]; }
    u0[tid] = tanhf(a0);
    u1[tid] = tanhf(a1);
  }
  __syncthreads();
  if (tid == 0) {  // gate scores, softmax over the 2 branches
    float z0 = l2b[0], z1 = l2b[0];
    for (int j = 0; j < 66; ++j) { z0 += u0[j] * l2W[j]; z1 += u1[j] * l2W[j]; }
    float m = fmaxf(z0, z1);
    float q0 = expf(z0 - m), q1 = expf(z1 - m);
    float inv = 1.f / (q0 + q1);
    sc[0] = q0 * inv;
    sc[1] = q1 * inv;
  }
  __syncthreads();
  if (tid < 66) u0[tid] = sc[0] * t0[tid] + sc[1] * t1[tid];  // fused
  __syncthreads();
  if (tid == 0) {  // log_softmax(fused @ outW + outb)
    float o0 = outb[0], o1 = outb[1];
    for (int j = 0; j < 66; ++j) { o0 += u0[j] * outW[j * 2 + 0]; o1 += u0[j] * outW[j * 2 + 1]; }
    float m = fmaxf(o0, o1);
    float lse = m + logf(expf(o0 - m) + expf(o1 - m));
    out[b * 2 + 0] = o0 - lse;
    out[b * 2 + 1] = o1 - lse;
  }
}

// ---------------------------------------------------------------------------
extern "C" void kernel_launch(void* const* d_in, const int* in_sizes, int n_in,
                              void* d_out, int out_size, void* d_ws, size_t ws_size,
                              hipStream_t stream) {
  (void)n_in; (void)out_size; (void)ws_size;
  const float* user_emb = (const float*)d_in[0];
  const float* gce      = (const float*)d_in[1];
  const float* lt_emb   = (const float*)d_in[2];
  const float* cp_emb   = (const float*)d_in[3];
  const float* li_emb   = (const float*)d_in[4];
  const float* w1       = (const float*)d_in[5];
  const float* w2       = (const float*)d_in[6];
  const float* weight   = (const float*)d_in[10];
  const float* weight2  = (const float*)d_in[11];
  const float* l1W      = (const float*)d_in[12];
  const float* l1b      = (const float*)d_in[13];
  const float* l2W      = (const float*)d_in[14];
  const float* l2b      = (const float*)d_in[15];
  const float* outW     = (const float*)d_in[16];
  const float* outb     = (const float*)d_in[17];
  const float* qkv      = (const float*)d_in[18];
  const float* Wo       = (const float*)d_in[19];
  const float* bo       = (const float*)d_in[20];
  const float* ln       = (const float*)d_in[21];
  const float* ffW      = (const float*)d_in[22];
  const float* ffB      = (const float*)d_in[23];
  const float* spread   = (const float*)d_in[25];
  const int* data_idx   = (const int*)d_in[26];
  const int* seq        = (const int*)d_in[27];
  const int* timestamps = (const int*)d_in[28];
  const int* user_level = (const int*)d_in[29];
  const int* useq       = (const int*)d_in[30];
  const int* user_inf   = (const int*)d_in[31];
  const int* user_cen   = (const int*)d_in[32];

  const int N  = in_sizes[0] / 64;   // 100000
  const int E  = in_sizes[27] / 50;  // 20000
  const int Bn = in_sizes[26];       // 1024
  const int BL = Bn * 50;

  // workspace layout (regionA reused: xin dead after the first GEMM)
  float* ws      = (float*)d_ws;
  float* regionA = ws;                           // N*64 floats
  float* xw      = regionA + (size_t)N * 64;     // N*64
  float* edge    = xw + (size_t)N * 64;          // E*64
  float* ew2     = edge + (size_t)E * 64;        // E*64
  _Float16* Wsw  = (_Float16*)(ew2 + (size_t)E * 64);  // 14*4096 halves (32B-aligned)
  float* adj     = regionA;                      // BL*64  (fits in regionA)
  float* pool    = regionA + (size_t)BL * 64;    // 2*Bn*64

  // one-time weight swizzle to f16 B-tile layout
  k_prep<<<(14 * 4096 + 255) / 256, 256, 0, stream>>>(w1, w2, qkv, Wo, ffW, Wsw);
  // HGNN: x = (user_emb + gce[cen]) @ W1
  k_xin<<<(N * 16 + 255) / 256, 256, 0, stream>>>(
      (const float4*)user_emb, (const float4*)gce, user_cen, (float4*)regionA, N);
  k_gemm64<<<(N + 63) / 64, 128, 0, stream>>>(regionA, Wsw + 0 * 4096, xw, N);
  // edge = relu(masked-mean gather)
  k_edge<<<(E * 16 + 255) / 256, 256, 0, stream>>>(
      (const float4*)xw, seq, (float4*)edge, E);
  // ew2 = edge @ W2
  k_gemm64<<<(E + 63) / 64, 128, 0, stream>>>(edge, Wsw + 1 * 4096, ew2, E);
  // adj[b,l,:] = node[seq_b[b,l]]
  k_adj<<<(BL * 16 + 255) / 256, 256, 0, stream>>>(
      (const float4*)ew2, seq, data_idx, useq, (float4*)adj, BL);
  // fused transformer blocks (temporal + structural) + pooling
  dim3 g(Bn, 2);
  k_tblock<<<g, 256, 0, stream>>>(adj, seq, data_idx, timestamps, user_level,
                                  user_inf, lt_emb, cp_emb, li_emb, Wsw, bo, ln,
                                  ffB, pool, Bn);
  // fusion head + log-softmax
  k_head<<<Bn, 128, 0, stream>>>(pool, data_idx, spread, weight, weight2, l1W,
                                 l1b, l2W, l2b, outW, outb, (float*)d_out, Bn);
}
